// AVWDCRNN_64561948393775
// MI455X (gfx1250) — matrix operand order repaired and checked by reading, not compiled, not run on hardware
//
#include <hip/hip_runtime.h>
#include <hip/hip_bf16.h>

// ---------------------------------------------------------------------------
// AVWDCRNN (RGSL/AGCRN 2-layer GRU graph RNN) for gfx1250.
// f16 WMMA (v_wmma_f32_16x16x32_f16) for all large matmuls, fp32 accumulate.
//   supports0 = I (free), supports1 = 0.5(S+L), supports2 = S^2 + L^2 - I.
//   All GEMM B-operands stored transposed so both A and B fragments load as
//   contiguous 128-bit LDS reads.  Global->LDS staging is double-buffered:
//   chunk k+1 loads overlap chunk k WMMAs, one barrier per K-chunk.
// ---------------------------------------------------------------------------

typedef _Float16 half_t;
typedef __attribute__((ext_vector_type(16))) _Float16 v16h;
typedef __attribute__((ext_vector_type(8)))  _Float16 h8;
typedef __attribute__((ext_vector_type(8)))  float    v8f;

constexpr int BB  = 64;   // batch
constexpr int TT  = 24;   // timesteps
constexpr int NN  = 512;  // nodes
constexpr int DIN = 2;
constexpr int HH  = 64;
constexpr int DD  = 10;
constexpr int KK  = 3;

#define DEV static __device__ __forceinline__

DEV v8f wmma(v16h a, v16h b, v8f c) {
  // 8 args: (neg_a, A, neg_b, B, c_mod, C, reuse_a, reuse_b)
  return __builtin_amdgcn_wmma_f32_16x16x32_f16(false, a, false, b,
                                                (short)0, c, false, false);
}

DEV v16h cat8(h8 a, h8 b) {
  return __builtin_shufflevector(a, b, 0, 1, 2, 3, 4, 5, 6, 7,
                                 8, 9, 10, 11, 12, 13, 14, 15);
}

// A fragment: 16x32 f16 tile in LDS, ld=32 (row-major).
// ISA 7.12.2 map: lane(0-15) row M=lane, halves -> K {hi*8..+7, 16+hi*8..+7}.
DEV v16h load_a_frag(const half_t* src) {
  int lane = threadIdx.x & 31;
  int lo = lane & 15, hi = lane >> 4;
  const h8* p = reinterpret_cast<const h8*>(src + lo * 32 + hi * 8);
  return cat8(p[0], p[2]);
}

// B fragment from a transposed-staged tile: 64(col) x 32(k), ld=32.
// column N=lo, halves -> K {hi*8..+7, 16+hi*8..+7} (matches A fragment).
DEV v16h load_bt_frag(const half_t* src) {
  int lane = threadIdx.x & 31;
  int lo = lane & 15, hi = lane >> 4;
  const h8* p = reinterpret_cast<const h8*>(src + lo * 32 + hi * 8);
  return cat8(p[0], p[2]);
}

// Combined staging of both 64x32 tiles (A + transposed-B), 128 threads.
// Each thread owns one contiguous 16-half row segment per tile.  All four
// global b128 loads are issued before any LDS store so they pipeline; the
// k+2 tiles are prefetched (global_prefetch_b8).
DEV void stage_both(half_t* sA, const half_t* gA, int lda,
                    half_t* sB, const half_t* gB, int ldb,
                    const half_t* pfA, const half_t* pfB) {
  int tid = threadIdx.x;
  int row = tid >> 1, col = (tid & 1) << 4;
  const h8* a = reinterpret_cast<const h8*>(&gA[(size_t)row * lda + col]);
  const h8* b = reinterpret_cast<const h8*>(&gB[(size_t)row * ldb + col]);
  h8 a0 = a[0], a1 = a[1];
  h8 b0 = b[0], b1 = b[1];
  if (pfA) __builtin_prefetch(&pfA[(size_t)row * lda + col], 0, 1);
  if (pfB) __builtin_prefetch(&pfB[(size_t)row * ldb + col], 0, 1);
  h8* da = reinterpret_cast<h8*>(&sA[row * 32 + col]);
  h8* db = reinterpret_cast<h8*>(&sB[row * 32 + col]);
  da[0] = a0; da[1] = a1;
  db[0] = b0; db[1] = b1;
}

// 4 waves; wave w computes rows [w*16, w*16+16) x 64 cols of the block tile.
DEV void mma_64x64(const half_t* sA, const half_t* sBt, v8f acc[4]) {
  int wave = threadIdx.x >> 5;
  v16h a = load_a_frag(sA + wave * 16 * 32);
#pragma unroll
  for (int j = 0; j < 4; ++j) {
    v16h b = load_bt_frag(sBt + j * 16 * 32);
    acc[j] = wmma(a, b, acc[j]);
  }
}

// Double-buffered K loop: C_tile += gA[64 x kmax] * gB^T[64 x kmax].
DEV void gemm_loop(const half_t* gA, int lda, const half_t* gB, int ldb,
                   int kmax, half_t (*sA)[64 * 32], half_t (*sB)[64 * 32],
                   v8f acc[4]) {
  stage_both(sA[0], gA, lda, sB[0], gB, ldb,
             kmax > 32 ? gA + 32 : nullptr, kmax > 32 ? gB + 32 : nullptr);
  __syncthreads();
  int buf = 0;
  for (int kk = 0; kk < kmax; kk += 32) {
    int nb = buf ^ 1;
    if (kk + 32 < kmax) {
      const half_t* pa = (kk + 64 < kmax) ? gA + kk + 64 : nullptr;
      const half_t* pb = (kk + 64 < kmax) ? gB + kk + 64 : nullptr;
      stage_both(sA[nb], gA + kk + 32, lda, sB[nb], gB + kk + 32, ldb, pa, pb);
    }
    mma_64x64(sA[buf], sB[buf], acc);
    __syncthreads();
    buf = nb;
  }
}

// ---------------------------------------------------------------------------
// S = softmax(relu(E E^T)) per row; emit f16 copies (plain + transposed) and
// A1 = 0.5(S+L).
// ---------------------------------------------------------------------------
__global__ __launch_bounds__(256) void k_graph(
    const float* __restrict__ E, const float* __restrict__ L,
    float* __restrict__ S, half_t* __restrict__ Sh,
    half_t* __restrict__ Lh, half_t* __restrict__ ShT,
    half_t* __restrict__ LhT, half_t* __restrict__ A1) {
  __shared__ float red[256];
  int n = blockIdx.x, tid = threadIdx.x;
  float en[DD];
#pragma unroll
  for (int d = 0; d < DD; ++d) en[d] = E[n * DD + d];
  float mx = -1e30f;
  for (int m = tid; m < NN; m += 256) {
    float v = 0.f;
#pragma unroll
    for (int d = 0; d < DD; ++d) v += en[d] * E[m * DD + d];
    v = fmaxf(v, 0.f);
    S[(size_t)n * NN + m] = v;
    mx = fmaxf(mx, v);
  }
  red[tid] = mx; __syncthreads();
  for (int s = 128; s > 0; s >>= 1) {
    if (tid < s) red[tid] = fmaxf(red[tid], red[tid + s]);
    __syncthreads();
  }
  mx = red[0]; __syncthreads();
  float sum = 0.f;
  for (int m = tid; m < NN; m += 256) {
    float e = __expf(S[(size_t)n * NN + m] - mx);
    S[(size_t)n * NN + m] = e;
    sum += e;
  }
  red[tid] = sum; __syncthreads();
  for (int s = 128; s > 0; s >>= 1) {
    if (tid < s) red[tid] += red[tid + s];
    __syncthreads();
  }
  float inv = 1.f / red[0];
  for (int m = tid; m < NN; m += 256) {
    float p = S[(size_t)n * NN + m] * inv;
    float l = L[(size_t)n * NN + m];
    S[(size_t)n * NN + m] = p;
    half_t ph = (half_t)p, lh = (half_t)l;
    Sh [(size_t)n * NN + m] = ph;
    Lh [(size_t)n * NN + m] = lh;
    ShT[(size_t)m * NN + n] = ph;
    LhT[(size_t)m * NN + n] = lh;
    A1 [(size_t)n * NN + m] = (half_t)(0.5f * (p + l));
  }
}

// ---------------------------------------------------------------------------
// A2 = Sh@Sh + Lh@Lh - I  (512x512, f16 out, f32 accumulate via WMMA).
// ---------------------------------------------------------------------------
__global__ __launch_bounds__(128) void k_gemm_sq(
    const half_t* __restrict__ Sh, const half_t* __restrict__ Lh,
    const half_t* __restrict__ ShT, const half_t* __restrict__ LhT,
    half_t* __restrict__ A2) {
  __shared__ __align__(16) half_t sA[2][64 * 32];
  __shared__ __align__(16) half_t sB[2][64 * 32];
  int row0 = blockIdx.y * 64, col0 = blockIdx.x * 64;
  v8f acc[4] = {};
  gemm_loop(Sh + (size_t)row0 * NN, NN, ShT + (size_t)col0 * NN, NN, NN,
            sA, sB, acc);
  gemm_loop(Lh + (size_t)row0 * NN, NN, LhT + (size_t)col0 * NN, NN, NN,
            sA, sB, acc);
  int lane = threadIdx.x & 31, wave = threadIdx.x >> 5;
  int lo = lane & 15, hi = lane >> 4;
#pragma unroll
  for (int j = 0; j < 4; ++j) {
    int col = col0 + j * 16 + lo;
#pragma unroll
    for (int r = 0; r < 8; ++r) {
      int n = row0 + wave * 16 + hi * 8 + r;
      float v = acc[j][r] - (n == col ? 1.f : 0.f);
      A2[(size_t)n * NN + col] = (half_t)v;
    }
  }
}

// ---------------------------------------------------------------------------
// Graph diffusion: P_k = A_k (512x512) @ X (512 x ncols), X given transposed
// as Xt[col][node]; result scattered into xg[n][b][kslot*Ci + c] (f16).
// blockIdx.z selects A1 (k=1) or A2 (k=2).
// ---------------------------------------------------------------------------
__global__ __launch_bounds__(128) void k_gemm_supports(
    const half_t* __restrict__ A1, const half_t* __restrict__ A2,
    const half_t* __restrict__ Xt, half_t* __restrict__ xg,
    int Ci, int kcpad) {
  __shared__ __align__(16) half_t sA[2][64 * 32];
  __shared__ __align__(16) half_t sB[2][64 * 32];
  const half_t* A = blockIdx.z ? A2 : A1;
  const int kslot = blockIdx.z ? 2 : 1;
  int row0 = blockIdx.y * 64, col0 = blockIdx.x * 64;
  v8f acc[4] = {};
  gemm_loop(A + (size_t)row0 * NN, NN, Xt + (size_t)col0 * NN, NN, NN,
            sA, sB, acc);
  int lane = threadIdx.x & 31, wave = threadIdx.x >> 5;
  int lo = lane & 15, hi = lane >> 4;
#pragma unroll
  for (int j = 0; j < 4; ++j) {
    int col = col0 + j * 16 + lo;
    int b = col / Ci, c = col - b * Ci;
#pragma unroll
    for (int r = 0; r < 8; ++r) {
      int n = row0 + wave * 16 + hi * 8 + r;
      xg[((size_t)n * BB + b) * kcpad + kslot * Ci + c] = (half_t)acc[j][r];
    }
  }
}

// ---------------------------------------------------------------------------
// Per-node GEMM: Y[64,O] = xg_n[64,kcpad] @ W_n[kcpad,O] + bias[n],
// W given transposed: Wt[n][o][k].
// mode 0 (gate):   sigmoid -> zr[b][n][2H]
// mode 1 (update): tanh -> hc; h = r*state + (1-r)*hc; write state & outputs.
// ---------------------------------------------------------------------------
__global__ __launch_bounds__(128) void k_node_gemm(
    const half_t* __restrict__ xg, const half_t* __restrict__ Wt,
    const float* __restrict__ bias, int kcpad, int Ofull, int mode,
    float* __restrict__ zr, float* __restrict__ state, int t, int layer,
    half_t* __restrict__ seq0, float* __restrict__ yout) {
  __shared__ __align__(16) half_t sA[2][64 * 32];
  __shared__ __align__(16) half_t sB[2][64 * 32];
  int n = blockIdx.x;
  int col0 = blockIdx.y * 64;
  const half_t* An  = xg + (size_t)n * BB * kcpad;
  const half_t* Wtn = Wt + (size_t)n * Ofull * kcpad + (size_t)col0 * kcpad;
  v8f acc[4] = {};
  gemm_loop(An, kcpad, Wtn, kcpad, kcpad, sA, sB, acc);
  int lane = threadIdx.x & 31, wave = threadIdx.x >> 5;
  int lo = lane & 15, hi = lane >> 4;
#pragma unroll
  for (int j = 0; j < 4; ++j) {
    int o = col0 + j * 16 + lo;
#pragma unroll
    for (int r = 0; r < 8; ++r) {
      int b = wave * 16 + hi * 8 + r;
      float v = acc[j][r] + bias[n * Ofull + o];
      if (mode == 0) {
        v = 1.f / (1.f + __expf(-v));
        zr[((size_t)b * NN + n) * (2 * HH) + o] = v;
      } else {
        float hc = tanhf(v);
        float rg = zr[((size_t)b * NN + n) * (2 * HH) + HH + o];
        float st = state[((size_t)b * NN + n) * HH + o];
        float h = rg * st + (1.f - rg) * hc;
        state[((size_t)b * NN + n) * HH + o] = h;
        if (layer == 0)
          seq0[(((size_t)t * BB + b) * NN + n) * HH + o] = (half_t)h;
        else
          yout[(((size_t)b * TT + t) * NN + n) * HH + o] = h;
      }
    }
  }
}

// ---------------------------------------------------------------------------
// Build Xt[(b,c)][n] = cat[b,n,c] (f16, transposed for GEMM-B) and the k=0
// slot of xg (supports[0] == I).  usez=1 -> candidate: state scaled by z.
// ---------------------------------------------------------------------------
__global__ void k_build_xmat(const float* __restrict__ x32,
                             const half_t* __restrict__ x16,
                             int Cin, int Ci, int kcpad, int t, int usez,
                             const float* __restrict__ state,
                             const float* __restrict__ zr,
                             half_t* __restrict__ Xt,
                             half_t* __restrict__ xg) {
  int ncols = BB * Ci;
  size_t total = (size_t)NN * ncols;
  size_t idx = (size_t)blockIdx.x * blockDim.x + threadIdx.x;
  if (idx >= total) return;
  int n = (int)(idx / ncols);
  int rem = (int)(idx % ncols);
  int b = rem / Ci, c = rem - b * Ci;
  float v;
  if (c < Cin) {
    if (x32) v = x32[(((size_t)b * TT + t) * NN + n) * Cin + c];        // layer0
    else     v = (float)x16[(((size_t)t * BB + b) * NN + n) * Cin + c]; // seq0
  } else {
    int cc = c - Cin;
    v = state[((size_t)b * NN + n) * HH + cc];
    if (usez) v *= zr[((size_t)b * NN + n) * (2 * HH) + cc];
  }
  half_t hv = (half_t)v;
  Xt[(size_t)rem * NN + n] = hv;
  xg[((size_t)n * BB + b) * kcpad + c] = hv;
}

// ---------------------------------------------------------------------------
// Per-node weights, TRANSPOSED: Wt[n][o][row] = sum_d E[n,d]*gw[d,k,i,o].
// Rows >= 3*Ci zero-padded so padded K-chunks contribute nothing.
// ---------------------------------------------------------------------------
__global__ void k_make_w(const float* __restrict__ E,
                         const float* __restrict__ gw,
                         half_t* __restrict__ Wout,
                         int Ci, int Ofull, int kcpad) {
  size_t total = (size_t)NN * kcpad * Ofull;
  size_t idx = (size_t)blockIdx.x * blockDim.x + threadIdx.x;
  if (idx >= total) return;
  int n = (int)(idx / ((size_t)kcpad * Ofull));
  int rem = (int)(idx % ((size_t)kcpad * Ofull));
  int o = rem / kcpad, row = rem - o * kcpad;   // Wt[n][o][row]
  float v = 0.f;
  if (row < KK * Ci) {
    int k = row / Ci, i = row - k * Ci;
    const float* g = gw + ((size_t)k * Ci + i) * Ofull + o;
    size_t dstride = (size_t)KK * Ci * Ofull;
#pragma unroll
    for (int d = 0; d < DD; ++d) v += E[n * DD + d] * g[d * dstride];
  }
  Wout[idx] = (half_t)v;
}

__global__ void k_make_b(const float* __restrict__ E,
                         const float* __restrict__ gb,
                         float* __restrict__ bout, int Ofull) {
  int idx = blockIdx.x * blockDim.x + threadIdx.x;
  if (idx >= NN * Ofull) return;
  int n = idx / Ofull, o = idx - n * Ofull;
  float v = 0.f;
#pragma unroll
  for (int d = 0; d < DD; ++d) v += E[n * DD + d] * gb[d * Ofull + o];
  bout[idx] = v;
}

// ---------------------------------------------------------------------------
extern "C" void kernel_launch(void* const* d_in, const int* in_sizes, int n_in,
                              void* d_out, int out_size, void* d_ws,
                              size_t ws_size, hipStream_t stream) {
  (void)in_sizes; (void)n_in; (void)out_size; (void)ws_size;
  const float* x   = (const float*)d_in[0];
  const float* ist = (const float*)d_in[1];
  const float* E   = (const float*)d_in[2];
  const float* L   = (const float*)d_in[3];
  const float* gw0 = (const float*)d_in[4];
  const float* gb0 = (const float*)d_in[5];
  const float* uw0 = (const float*)d_in[6];
  const float* ub0 = (const float*)d_in[7];
  const float* gw1 = (const float*)d_in[8];
  const float* gb1 = (const float*)d_in[9];
  const float* uw1 = (const float*)d_in[10];
  const float* ub1 = (const float*)d_in[11];
  float* y = (float*)d_out;
  float* lasts = y + (size_t)BB * TT * NN * HH;

  char* ws = (char*)d_ws;
  size_t off = 0;
  auto alloc = [&](size_t bytes) -> void* {
    void* p = ws + off;
    off = (off + bytes + 255) & ~(size_t)255;
    return p;
  };
  const int KCP0 = 224;  // pad 3*66=198 -> 7*32
  const int KCP1 = 384;  // 3*128     -> 12*32
  float*  S    = (float*)  alloc((size_t)NN * NN * 4);
  half_t* Sh   = (half_t*) alloc((size_t)NN * NN * 2);
  half_t* Lh   = (half_t*) alloc((size_t)NN * NN * 2);
  half_t* ShT  = (half_t*) alloc((size_t)NN * NN * 2);
  half_t* LhT  = (half_t*) alloc((size_t)NN * NN * 2);
  half_t* A1   = (half_t*) alloc((size_t)NN * NN * 2);
  half_t* A2   = (half_t*) alloc((size_t)NN * NN * 2);
  half_t* Wg0  = (half_t*) alloc((size_t)NN * KCP0 * 128 * 2);
  half_t* Wu0  = (half_t*) alloc((size_t)NN * KCP0 * 64 * 2);
  half_t* Wg1  = (half_t*) alloc((size_t)NN * KCP1 * 128 * 2);
  half_t* Wu1  = (half_t*) alloc((size_t)NN * KCP1 * 64 * 2);
  float*  bg0  = (float*)  alloc((size_t)NN * 128 * 4);
  float*  bu0  = (float*)  alloc((size_t)NN * 64 * 4);
  float*  bg1  = (float*)  alloc((size_t)NN * 128 * 4);
  float*  bu1  = (float*)  alloc((size_t)NN * 64 * 4);
  half_t* Xt   = (half_t*) alloc((size_t)NN * BB * 128 * 2);
  half_t* xg   = (half_t*) alloc((size_t)NN * BB * KCP1 * 2);
  float*  zr   = (float*)  alloc((size_t)BB * NN * 2 * HH * 4);
  float*  st   = (float*)  alloc((size_t)BB * NN * HH * 4);
  half_t* seq0 = (half_t*) alloc((size_t)TT * BB * NN * HH * 2);

  // Graph setup
  k_graph<<<NN, 256, 0, stream>>>(E, L, S, Sh, Lh, ShT, LhT, A1);
  k_gemm_sq<<<dim3(NN / 64, NN / 64), 128, 0, stream>>>(Sh, Lh, ShT, LhT, A2);

  // Per-node weights / biases (time-invariant)
  auto launch_w = [&](const float* g, half_t* Wout, int Ci, int O, int kcp) {
    size_t total = (size_t)NN * kcp * O;
    k_make_w<<<(unsigned)((total + 255) / 256), 256, 0, stream>>>(E, g, Wout,
                                                                 Ci, O, kcp);
  };
  launch_w(gw0, Wg0, 66, 128, KCP0);
  launch_w(uw0, Wu0, 66, 64, KCP0);
  launch_w(gw1, Wg1, 128, 128, KCP1);
  launch_w(uw1, Wu1, 128, 64, KCP1);
  k_make_b<<<(NN * 128 + 255) / 256, 256, 0, stream>>>(E, gb0, bg0, 128);
  k_make_b<<<(NN * 64 + 255) / 256, 256, 0, stream>>>(E, ub0, bu0, 64);
  k_make_b<<<(NN * 128 + 255) / 256, 256, 0, stream>>>(E, gb1, bg1, 128);
  k_make_b<<<(NN * 64 + 255) / 256, 256, 0, stream>>>(E, ub1, bu1, 64);

  for (int layer = 0; layer < 2; ++layer) {
    const int Cin   = layer ? HH : DIN;
    const int Ci    = Cin + HH;          // 66 / 128
    const int kcp   = layer ? KCP1 : KCP0;
    const int ncols = BB * Ci;           // 4224 / 8192 (both /64 exact)
    half_t* Wg = layer ? Wg1 : Wg0;
    half_t* Wu = layer ? Wu1 : Wu0;
    float*  bg = layer ? bg1 : bg0;
    float*  bu = layer ? bu1 : bu0;
    hipMemcpyAsync(st, ist + (size_t)layer * BB * NN * HH,
                   (size_t)BB * NN * HH * 4, hipMemcpyDeviceToDevice, stream);
    for (int t = 0; t < TT; ++t) {
      const float*  x32 = layer ? nullptr : x;
      const half_t* x16 = layer ? seq0 : nullptr;
      size_t total = (size_t)NN * ncols;
      unsigned bblk = (unsigned)((total + 255) / 256);
      // gate: cat = [x_t, state]
      k_build_xmat<<<bblk, 256, 0, stream>>>(x32, x16, Cin, Ci, kcp, t, 0, st,
                                             zr, Xt, xg);
      k_gemm_supports<<<dim3(ncols / 64, NN / 64, 2), 128, 0, stream>>>(
          A1, A2, Xt, xg, Ci, kcp);
      k_node_gemm<<<dim3(NN, 2), 128, 0, stream>>>(xg, Wg, bg, kcp, 128, 0, zr,
                                                   st, t, layer, seq0, y);
      // candidate: cand = [x_t, z*state]
      k_build_xmat<<<bblk, 256, 0, stream>>>(x32, x16, Cin, Ci, kcp, t, 1, st,
                                             zr, Xt, xg);
      k_gemm_supports<<<dim3(ncols / 64, NN / 64, 2), 128, 0, stream>>>(
          A1, A2, Xt, xg, Ci, kcp);
      k_node_gemm<<<dim3(NN, 1), 128, 0, stream>>>(xg, Wu, bu, kcp, 64, 1, zr,
                                                   st, t, layer, seq0, y);
    }
    hipMemcpyAsync(lasts + (size_t)layer * BB * NN * HH, st,
                   (size_t)BB * NN * HH * 4, hipMemcpyDeviceToDevice, stream);
  }
}
